// GraphicLaplacianAttention_11295763988903
// MI455X (gfx1250) — matrix operand order repaired and checked by reading, not compiled, not run on hardware
//
#include <hip/hip_runtime.h>

typedef float v2f __attribute__((ext_vector_type(2)));
typedef float v8f __attribute__((ext_vector_type(8)));
typedef __bf16 v16bf __attribute__((ext_vector_type(16)));

#define NB 16
#define NC 128
#define NH 128
#define NW 128
#define NHW (NH * NW)
#define XS32_STR 36  // f32 tile row stride (words): float4-aligned, banks OK
#define PKB_STR 36   // packed bf16 pair-row stride (words): halves 16 banks apart

union ABu {
  v16bf bf;
  unsigned u[8];
};

static __device__ inline unsigned pack2(__bf16 lo, __bf16 hi) {
  union { __bf16 b[2]; unsigned u; } u_;
  u_.b[0] = lo;
  u_.b[1] = hi;
  return u_.u;
}

static __device__ inline v8f wmma_bf16(v16bf a, v16bf b, v8f c) {
  return __builtin_amdgcn_wmma_f32_16x16x32_bf16(false, a, false, b, (short)0, c,
                                                 false, false);
}

// byte offset of a generic pointer within the workgroup LDS segment
static __device__ inline unsigned lds_off_of(const void* p) {
  return (unsigned)(unsigned long long)(__attribute__((address_space(3))) const char*)p;
}

// ---------------------------------------------------------------------------
// Stage A: per-(b,c) plane: channel max (-> sigmoid) and 16x16 block-mean pool
// ---------------------------------------------------------------------------
__global__ __launch_bounds__(256) void k_pool_max(const float* __restrict__ x,
                                                  float* __restrict__ cmax,
                                                  float* __restrict__ xp) {
  const int bc = blockIdx.x;  // b*NC + c
  const int t = threadIdx.x;  // 256 threads = 16x16 pooled cells
  const int py = t >> 4, px = t & 15;
  const float* base = x + (size_t)bc * NHW;
  float mx = -3.4e38f;
  float sum = 0.f;
  #pragma unroll
  for (int rr = 0; rr < 8; ++rr) {
    const float4* r4 = (const float4*)(base + (py * 8 + rr) * NW + px * 8);
    float4 u = r4[0], v = r4[1];
    sum += u.x + u.y + u.z + u.w + v.x + v.y + v.z + v.w;
    mx = fmaxf(mx, fmaxf(fmaxf(u.x, u.y), fmaxf(u.z, u.w)));
    mx = fmaxf(mx, fmaxf(fmaxf(v.x, v.y), fmaxf(v.z, v.w)));
  }
  xp[(size_t)bc * 256 + t] = sum * (1.0f / 64.0f);
  __shared__ float red[256];
  red[t] = mx;
  __syncthreads();
  for (int s = 128; s > 0; s >>= 1) {
    if (t < s) red[t] = fmaxf(red[t], red[t + s]);
    __syncthreads();
  }
  if (t == 0) cmax[bc] = 1.0f / (1.0f + expf(-red[0]));
}

// ---------------------------------------------------------------------------
// Stage B: per-batch 128x128 Laplacian matrix, emitted as packed bf16 hi/lo
// in WMMA A-fragment word order: word j/2 of row i = {bf16(l[j]), bf16(l[j+1])}
// ---------------------------------------------------------------------------
__global__ __launch_bounds__(128) void k_lap(const float* __restrict__ cmax,
                                             unsigned* __restrict__ laph,
                                             unsigned* __restrict__ lapl) {
  const int b = blockIdx.x;
  const int t = threadIdx.x;  // 0..127
  __shared__ float cm[NC], inrm[NC], sig[NC], rs[NC], cs[NC], red[NC];
  cm[t] = cmax[b * NC + t];
  __syncthreads();
  // column L2 norm of s[:,t], s[i][j] = cmax[i] + 1 - cmax[j]
  const float cj = cm[t];
  float acc = 0.f;
  for (int i = 0; i < NC; ++i) { float s = cm[i] + 1.0f - cj; acc += s * s; }
  const float nj = fmaxf(sqrtf(acc), 1e-12f);
  inrm[t] = 1.0f / nj;
  red[t] = 1.0f / nj;  // diag of d_c (s[t][t] == 1)
  __syncthreads();
  for (int s = 64; s > 0; s >>= 1) {
    if (t < s) red[t] = fminf(red[t], red[t + s]);
    __syncthreads();
  }
  const float r = fminf(red[0], 1.0f);
  // sigma (row min of d_c), store 1/(2*sigma^2 + eps)
  const float ci = cm[t];
  float mn = 3.4e38f;
  for (int j = 0; j < NC; ++j) mn = fminf(mn, (ci + 1.0f - cm[j]) * inrm[j]);
  const float sgm = fmaxf(mn, 0.5f);
  sig[t] = 1.0f / (2.0f * sgm * sgm + 1e-8f);
  __syncthreads();
  // row sum of g^2 (row t uses sig[t])
  const float inv2 = sig[t];
  float rsum = 0.f;
  for (int j = 0; j < NC; ++j) {
    float d = fabsf(ci - cm[j]) - r;
    float g = expf(-d * d * inv2);
    rsum += g * g;
  }
  rs[t] = rsum;
  // column sum of g^2 (g[i][t] uses sig[i])
  float csum = 0.f;
  for (int i = 0; i < NC; ++i) {
    float d = fabsf(cm[i] - ci) - r;
    float g = expf(-d * d * sig[i]);
    csum += g * g;
  }
  cs[t] = csum;
  __syncthreads();
  unsigned* hrow = laph + ((size_t)(b * NC + t)) * (NC / 2);
  unsigned* lrow = lapl + ((size_t)(b * NC + t)) * (NC / 2);
  for (int j = 0; j < NC; j += 2) {
    float a0 = 0.5f * (sqrtf(rs[t] + cs[j]) + sqrtf(rs[j] + cs[t]));
    float a1 = 0.5f * (sqrtf(rs[t] + cs[j + 1]) + sqrtf(rs[j + 1] + cs[t]));
    float l0 = ci * (((j == t) ? 1.0f : 0.0f) + a0);
    float l1 = ci * (((j + 1 == t) ? 1.0f : 0.0f) + a1);
    __bf16 h0 = (__bf16)l0, h1 = (__bf16)l1;
    float r0 = l0 - (float)h0, r1 = l1 - (float)h1;
    hrow[j >> 1] = pack2(h0, h1);
    lrow[j >> 1] = pack2((__bf16)r0, (__bf16)r1);
  }
}

// ---------------------------------------------------------------------------
// Stage C: per-batch 256x256 spatial softmax attention "a_s"
// ---------------------------------------------------------------------------
__global__ __launch_bounds__(256) void k_as(const float* __restrict__ xp,
                                            float* __restrict__ as_) {
  const int b = blockIdx.x;
  const int p = threadIdx.x;  // 0..255
  __shared__ float v[256], sg[256], red[256];
  const float* xpb = xp + (size_t)b * NC * 256;
  float acc = 0.f;
  for (int c = 0; c < NC; ++c) acc += xpb[c * 256 + p];
  const float vp = acc * (1.0f / NC);
  v[p] = vp;
  red[p] = vp * vp;
  __syncthreads();
  for (int s = 128; s > 0; s >>= 1) {
    if (p < s) red[p] += red[p + s];
    __syncthreads();
  }
  const float vnorm = sqrtf(red[0]);
  __syncthreads();
  const float denom = fmaxf(fabsf(vp) * vnorm, 1e-12f);
  float mn = 3.4e38f;
  for (int q = 0; q < 256; ++q) mn = fminf(mn, vp * v[q] / denom);
  const float sgm = fmaxf(mn, 0.5f);
  sg[p] = 1.0f / (2.0f * sgm * sgm + 1e-8f);
  red[p] = vp * vp / denom;  // diag of d_s
  __syncthreads();
  for (int s = 128; s > 0; s >>= 1) {
    if (p < s) red[p] = fminf(red[p], red[p + s]);
    __syncthreads();
  }
  const float r = fminf(red[0], 1.0f);
  __syncthreads();
  // row softmax over q (exponents <= 0, no max-shift needed)
  float ssum = 0.f;
  for (int q = 0; q < 256; ++q) {
    float d = fabsf(vp - v[q]) - r;
    ssum += expf(-d * d * sg[q]);
  }
  const float inv = 1.0f / ssum;
  float* row = as_ + ((size_t)(b * 256 + p)) * 256;
  for (int q = 0; q < 256; ++q) {
    float d = fabsf(vp - v[q]) - r;
    row[q] = expf(-d * d * sg[q]) * inv;
  }
}

// ---------------------------------------------------------------------------
// Stage D: enh[b] = xp[b] (128x256) @ a_s[b] (256x256), WMMA f32 16x16x4
// ---------------------------------------------------------------------------
__global__ __launch_bounds__(256) void k_enh(const float* __restrict__ xp,
                                             const float* __restrict__ as_,
                                             float* __restrict__ enh) {
  const int b = blockIdx.y;
  const int n0 = blockIdx.x * 64;
  const int wave = threadIdx.x >> 5;
  const int lane = threadIdx.x & 31;
  const int m0 = wave * 16;
  const int lhalf = lane >> 4, l16 = lane & 15;
  const float* Arow = xp + ((size_t)(b * NC + m0 + l16)) * 256;
  v8f acc[4];
  #pragma unroll
  for (int j = 0; j < 4; ++j) { v8f z = {0, 0, 0, 0, 0, 0, 0, 0}; acc[j] = z; }
  for (int kk = 0; kk < 256; kk += 4) {
    const int kb = kk + (lhalf << 1);
    v2f a;
    a.x = Arow[kb];
    a.y = Arow[kb + 1];
    #pragma unroll
    for (int j = 0; j < 4; ++j) {
      const int col = n0 + 16 * j + l16;
      v2f bf;
      bf.x = as_[((size_t)(b * 256 + kb)) * 256 + col];
      bf.y = as_[((size_t)(b * 256 + kb + 1)) * 256 + col];
      acc[j] = __builtin_amdgcn_wmma_f32_16x16x4_f32(false, a, false, bf,
                                                     (short)0, acc[j], false, false);
    }
  }
  #pragma unroll
  for (int j = 0; j < 4; ++j) {
    const int col = n0 + 16 * j + l16;
    #pragma unroll
    for (int g = 0; g < 8; ++g) {
      const int m = m0 + g + 8 * lhalf;
      enh[((size_t)(b * NC + m)) * 256 + col] = acc[j][g];
    }
  }
}

// ---------------------------------------------------------------------------
// Stage E (fused): async-to-LDS staging + split-bf16 WMMA x_c tile
// + bilinear upsample + final blend.  grid (NH*4, NB): bx = h*4 + 32-col tile
// ---------------------------------------------------------------------------
__global__ __launch_bounds__(256) void k_main(const float* __restrict__ x,
                                              const unsigned* __restrict__ laph,
                                              const unsigned* __restrict__ lapl,
                                              const float* __restrict__ enh,
                                              float* __restrict__ out) {
  const int b = blockIdx.y;
  const int h = blockIdx.x >> 2;
  const int w0 = (blockIdx.x & 3) * 32;
  const int t = threadIdx.x;
  __shared__ float xs[NC * XS32_STR];     // x[b,:,h,w0:w0+32] fp32   (18432 B)
  __shared__ unsigned xh[64 * PKB_STR];   // bf16 hi, {k,k+1} packed  ( 9216 B)
  __shared__ unsigned xl[64 * PKB_STR];   // bf16 lo, {k,k+1} packed  ( 9216 B)
  __shared__ float rb[NC * 16];           // enh y-blended at this h  ( 8192 B)

  // stage f32 tile via async DMA (ASYNCcnt path): 128 rows x 32 cols,
  // 4x b128 per lane; overlaps with the rb computation below
  const float* src = x + ((size_t)b * NC) * NHW + h * NW + w0;
  #pragma unroll
  for (int it = 0; it < 4; ++it) {
    const int idx = t + it * 256;  // 0..1023 float4 slots
    const int k = idx >> 3;
    const int c4 = (idx & 7) * 4;
    const unsigned loff = lds_off_of(&xs[k * XS32_STR + c4]);
    const unsigned long long ga =
        (unsigned long long)(src + (size_t)k * NHW + c4);
    asm volatile("global_load_async_to_lds_b128 %0, %1, off"
                 :
                 : "v"(loff), "v"(ga)
                 : "memory");
  }
  // y-direction bilinear blend of enh (half-pixel centers, scale 8)
  const float sy = (h + 0.5f) * 0.125f - 0.5f;
  const float fy = sy - floorf(sy);
  int iy0 = (int)floorf(sy);
  int iy1 = iy0 + 1;
  iy0 = min(max(iy0, 0), 15);
  iy1 = min(max(iy1, 0), 15);
  {
    const float* eb = enh + ((size_t)b * NC) * 256;
    #pragma unroll
    for (int it = 0; it < 8; ++it) {
      const int idx = t + it * 256;  // 2048 = 128 channels x 16 src cols
      const int c = idx >> 4;
      const int xc = idx & 15;
      const float e0 = eb[(size_t)c * 256 + iy0 * 16 + xc];
      const float e1 = eb[(size_t)c * 256 + iy1 * 16 + xc];
      rb[idx] = e0 + fy * (e1 - e0);
    }
  }
  asm volatile("s_wait_asynccnt 0x0" ::: "memory");
  __syncthreads();
  // pack split-bf16 pair rows: word = {bf16(x[2p][n]), bf16(x[2p+1][n])}
  #pragma unroll
  for (int it = 0; it < 8; ++it) {
    const int idx = t + it * 256;  // 0..2047
    const int p = idx >> 5;        // pair row (k = 2p, 2p+1)
    const int n = idx & 31;
    const float a0 = xs[(2 * p) * XS32_STR + n];
    const float a1 = xs[(2 * p + 1) * XS32_STR + n];
    const __bf16 h0 = (__bf16)a0, h1 = (__bf16)a1;
    const float r0 = a0 - (float)h0, r1 = a1 - (float)h1;
    xh[p * PKB_STR + n] = pack2(h0, h1);
    xl[p * PKB_STR + n] = pack2((__bf16)r0, (__bf16)r1);
  }
  __syncthreads();

  const int wave = t >> 5, lane = t & 31;
  const int m0 = wave * 16;
  const int lhalf = lane >> 4, l16 = lane & 15;
  const unsigned* AH = laph + ((size_t)(b * NC + m0 + l16)) * (NC / 2);
  const unsigned* AL = lapl + ((size_t)(b * NC + m0 + l16)) * (NC / 2);
  v8f acc[2];
  #pragma unroll
  for (int j = 0; j < 2; ++j) { v8f z = {0, 0, 0, 0, 0, 0, 0, 0}; acc[j] = z; }

  for (int kk = 0; kk < NC; kk += 32) {
    const int kw = kk >> 1;  // pair-row base
    // A-fragment: VGPR v holds K = (v>=4?16:0) + (v&3)*2 (+8 for upper lanes)
    ABu ah, al;
    #pragma unroll
    for (int v = 0; v < 8; ++v) {
      const int wi = kw + ((v >> 2) << 3) + (v & 3) + (lhalf << 2);
      ah.u[v] = AH[wi];
      al.u[v] = AL[wi];
    }
    #pragma unroll
    for (int j = 0; j < 2; ++j) {
      const int colw = 16 * j + l16;
      // B-fragment: VGPR v holds K = (v>=4?16:0) + (v&3)*2 (+8 for upper lanes)
      ABu bh, bl;
      #pragma unroll
      for (int v = 0; v < 8; ++v) {
        const int wi = (kw + ((v >> 2) << 3) + (v & 3) + (lhalf << 2)) * PKB_STR + colw;
        bh.u[v] = xh[wi];
        bl.u[v] = xl[wi];
      }
      acc[j] = wmma_bf16(ah.bf, bh.bf, acc[j]);
      acc[j] = wmma_bf16(ah.bf, bl.bf, acc[j]);
      acc[j] = wmma_bf16(al.bf, bh.bf, acc[j]);
    }
  }

  // epilogue: out = x + (1/3)*x_c + (2/3)*upsample(enh)
  const float kc = 1.0f / 3.0f, ks = 2.0f / 3.0f;
  #pragma unroll
  for (int j = 0; j < 2; ++j) {
    const int w = w0 + 16 * j + l16;
    const float sxf = (w + 0.5f) * 0.125f - 0.5f;
    const float fx = sxf - floorf(sxf);
    int ix0 = (int)floorf(sxf);
    int ix1 = ix0 + 1;
    ix0 = min(max(ix0, 0), 15);
    ix1 = min(max(ix1, 0), 15);
    #pragma unroll
    for (int g = 0; g < 8; ++g) {
      const int m = m0 + g + 8 * lhalf;
      const float s0 = rb[m * 16 + ix0];
      const float s1 = rb[m * 16 + ix1];
      const float sv = s0 + fx * (s1 - s0);
      const float xv = xs[m * XS32_STR + (w - w0)];
      out[((size_t)(b * NC + m)) * NHW + (size_t)h * NW + w] =
          xv + kc * acc[j][g] + ks * sv;
    }
  }
}

// ---------------------------------------------------------------------------
extern "C" void kernel_launch(void* const* d_in, const int* in_sizes, int n_in,
                              void* d_out, int out_size, void* d_ws, size_t ws_size,
                              hipStream_t stream) {
  (void)in_sizes; (void)n_in; (void)out_size; (void)ws_size;
  const float* x = (const float*)d_in[0];
  float* out = (float*)d_out;

  float* cmax = (float*)d_ws;                        // NB*NC        = 2048 f
  float* xp = cmax + NB * NC;                        // NB*NC*256    = 524288 f
  float* as_ = xp + (size_t)NB * NC * 256;           // NB*256*256   = 1048576 f
  float* enh = as_ + (size_t)NB * 256 * 256;         // NB*NC*256    = 524288 f
  unsigned* laph = (unsigned*)(enh + (size_t)NB * NC * 256);  // NB*NC*64 u32
  unsigned* lapl = laph + (size_t)NB * NC * (NC / 2);         // NB*NC*64 u32

  k_pool_max<<<dim3(NB * NC), dim3(256), 0, stream>>>(x, cmax, xp);
  k_lap<<<dim3(NB), dim3(128), 0, stream>>>(cmax, laph, lapl);
  k_as<<<dim3(NB), dim3(256), 0, stream>>>(xp, as_);
  k_enh<<<dim3(4, NB), dim3(256), 0, stream>>>(xp, as_, enh);
  k_main<<<dim3(NH * 4, NB), dim3(256), 0, stream>>>(x, laph, lapl, enh, out);
}
